// GENE_41592463294750
// MI455X (gfx1250) — matrix operand order repaired and checked by reading, not compiled, not run on hardware
//
#include <hip/hip_runtime.h>
#include <hip/hip_bf16.h>
#include <math.h>

#define NEG_INF (-9e15f)
#define LRELU_ALPHA 0.2f

typedef __bf16 bf16;
typedef __attribute__((ext_vector_type(16))) __bf16 v16bf;
typedef __attribute__((ext_vector_type(8)))  __bf16 v8bf;
typedef __attribute__((ext_vector_type(8)))  float  v8f;

static __device__ __forceinline__ v16bf cat8(v8bf lo, v8bf hi) {
  v16bf r;
#pragma unroll
  for (int i = 0; i < 8; ++i) { r[i] = lo[i]; r[8 + i] = hi[i]; }
  return r;
}
static __device__ __forceinline__ float lrelu(float x)   { return x >= 0.f ? x : LRELU_ALPHA * x; }
static __device__ __forceinline__ float elu1(float x)    { return x > 0.f ? x : (__expf(x) - 1.f); }
static __device__ __forceinline__ float sigmoidf(float x){ return 1.f / (1.f + __expf(-x)); }

// ---------------------------------------------------------------------------
// fp32 -> bf16 copy (row-major, A-operand role)
// ---------------------------------------------------------------------------
__global__ void k_f32_to_bf16(const float* __restrict__ in, bf16* __restrict__ out, int n) {
  int i = blockIdx.x * blockDim.x + threadIdx.x;
  if (i < n) out[i] = (bf16)in[i];
}

// fp32 [R][C] -> bf16 [C][R]  (B-operand role: stored transposed so lane K-gathers are contiguous)
__global__ void k_transpose_to_bf16(const float* __restrict__ in, bf16* __restrict__ out, int R, int C) {
  int i = blockIdx.x * blockDim.x + threadIdx.x;
  if (i >= R * C) return;
  int r = i / C, c = i % C;
  out[(size_t)c * R + r] = (bf16)in[(size_t)r * C + c];
}

// ---------------------------------------------------------------------------
// Generic WMMA GEMM: C[M][Nc] = A[M][K] * B[K][Nc], with B given as BT[Nc][K].
// One wave computes a 16x16 tile; K-loop of v_wmma_f32_16x16x32_bf16.
// Epilogue can emit fp32 row-major, bf16 row-major, bf16 transposed.
// ---------------------------------------------------------------------------
__global__ void k_gemm_wmma_bf16(const bf16* __restrict__ A, const bf16* __restrict__ BT,
                                 float* __restrict__ Cf32, bf16* __restrict__ Cbf_rm,
                                 bf16* __restrict__ Cbf_tr, int M, int K, int Nc) {
  int wave = (blockIdx.x * blockDim.x + threadIdx.x) >> 5;
  int lane = threadIdx.x & 31;
  int tilesN = Nc >> 4;
  int tm = wave / tilesN, tn = wave % tilesN;
  if (tm * 16 >= M) return;
  int half = lane >> 4, l16 = lane & 15;
  const bf16* arow = A  + (size_t)(tm * 16 + l16) * K;   // A: lane holds row m = lane%16
  const bf16* brow = BT + (size_t)(tn * 16 + l16) * K;   // B: lane holds col n = lane%16
  v8f c = {};
  for (int k0 = 0; k0 < K; k0 += 32) {
    // A 16x32 bf16 layout: K-chunks [half*8, half*8+8) and [16+half*8, ...)
    v16bf a = cat8(*(const v8bf*)(arow + k0 + half * 8),
                   *(const v8bf*)(arow + k0 + 16 + half * 8));
    // B 32x16 bf16 layout: K-chunks [half*8, half*8+8) and [16+half*8, ...)
    v16bf b = cat8(*(const v8bf*)(brow + k0 + half * 8),
                   *(const v8bf*)(brow + k0 + 16 + half * 8));
    c = __builtin_amdgcn_wmma_f32_16x16x32_bf16(false, a, false, b, (short)0, c, false, false);
  }
  int n = tn * 16 + l16;
#pragma unroll
  for (int r = 0; r < 8; ++r) {
    int m = tm * 16 + r + half * 8;           // C/D layout: VGPR r -> row r + half*8
    float v = c[r];
    if (Cf32)   Cf32  [(size_t)m * Nc + n] = v;
    if (Cbf_rm) Cbf_rm[(size_t)m * Nc + n] = (bf16)v;
    if (Cbf_tr) Cbf_tr[(size_t)n * M  + m] = (bf16)v;
  }
}

// ---------------------------------------------------------------------------
// f1 = Wh @ a[:dout], f2 = Wh @ a[dout:]  (one thread per row)
// ---------------------------------------------------------------------------
__global__ void k_compute_f(const float* __restrict__ Wh, const float* __restrict__ a,
                            float* __restrict__ f1, float* __restrict__ f2, int N, int dout) {
  int i = blockIdx.x * blockDim.x + threadIdx.x;
  if (i >= N) return;
  const float* row = Wh + (size_t)i * dout;
  float s1 = 0.f, s2 = 0.f;
  for (int k = 0; k < dout; ++k) {
    float w = row[k];
    s1 += w * a[k];
    s2 += w * a[dout + k];
  }
  f1[i] = s1; f2[i] = s2;
}

// ---------------------------------------------------------------------------
// Per-row online masked-softmax stats over one adj row:
//   m_i = max_j e_ij,  s_i = sum_j exp(e_ij - m_i),  e_ij = adj>0 ? lrelu(f1_i+f2_j) : NEG_INF
// NEG_INF is finite (-9e15) so the empty-row case collapses to exp(0)=1 terms,
// reproducing jax's uniform-softmax behavior exactly. One block (256 thr) per row.
// ---------------------------------------------------------------------------
__global__ void k_row_stats(const float* __restrict__ adj, const float* __restrict__ f1,
                            const float* __restrict__ f2, float* __restrict__ mrow,
                            float* __restrict__ srow, float* __restrict__ rowmask, int N) {
  int i = blockIdx.x, tid = threadIdx.x;
  const float* arow = adj + (size_t)i * N;
  float fi = f1[i];
  float lm = NEG_INF, ls = 0.f, any = 0.f;
  for (int j = tid; j < N; j += blockDim.x) {
    if (j + 2048 < N) __builtin_prefetch(arow + j + 2048, 0, 0);   // global_prefetch_b8
    float av = arow[j];
    float e = (av > 0.f) ? lrelu(fi + f2[j]) : NEG_INF;
    any = fmaxf(any, av);
    if (e > lm) { ls = ls * __expf(lm - e) + 1.f; lm = e; }
    else        { ls += __expf(e - lm); }
  }
  __shared__ float rm[256], rs[256], ra[256];
  rm[tid] = lm; rs[tid] = ls; ra[tid] = any;
  __syncthreads();
  for (int off = 128; off > 0; off >>= 1) {
    if (tid < off) {
      float m1 = rm[tid], s1 = rs[tid], m2 = rm[tid + off], s2 = rs[tid + off];
      float mm = fmaxf(m1, m2);
      rs[tid] = s1 * __expf(m1 - mm) + s2 * __expf(m2 - mm);
      rm[tid] = mm;
      ra[tid] = fmaxf(ra[tid], ra[tid + off]);
    }
    __syncthreads();
  }
  if (tid == 0) {
    mrow[i] = rm[0];
    srow[i] = rs[0];
    if (rowmask) rowmask[i] = (ra[0] > 0.f) ? 1.f : 0.f;
  }
}

// ---------------------------------------------------------------------------
// Fused attention construction + (att @ Wh) WMMA GEMM.
// Block = 16-row strip; blockDim = 2*dout (one wave per 16 output columns).
// Each K-step: cooperatively build the 16x32 att tile from adj/f1/f2/m/s,
// optionally stream it to the `attention` output, stash bf16 copy in LDS
// (A-operand), WMMA against WhT (B-operand, transposed -> contiguous loads).
// ---------------------------------------------------------------------------
__global__ void k_gat_att_gemm(const float* __restrict__ adj, const float* __restrict__ f1,
                               const float* __restrict__ f2, const float* __restrict__ mrow,
                               const float* __restrict__ srow, const bf16* __restrict__ WhT,
                               float* __restrict__ attOut, float* __restrict__ Hout,
                               bf16* __restrict__ Hbf, int N, int dout, int actElu) {
  __shared__ __align__(32) bf16 attbf[16 * 32];
  __shared__ float sF1[16], sM[16], sIS[16];
  int i0 = blockIdx.x * 16;
  int tid = threadIdx.x;
  if (tid < 16) {
    sF1[tid] = f1[i0 + tid];
    sM[tid]  = mrow[i0 + tid];
    sIS[tid] = 1.f / srow[i0 + tid];
  }
  __syncthreads();

  int lane = tid & 31, wv = tid >> 5;
  int half = lane >> 4, l16 = lane & 15;
  int col0 = wv * 16;
  const bf16* brow = WhT + (size_t)(col0 + l16) * N;
  int per = 512 / blockDim.x;                 // att-tile elements per thread
  v8f c = {};

  for (int j = 0; j < N; j += 32) {
    for (int t = 0; t < per; ++t) {
      int idx = t * blockDim.x + tid;         // idx = r*32 + jj
      int r = idx >> 5, jj = idx & 31;
      size_t gi = (size_t)(i0 + r) * N + j + jj;
      __builtin_prefetch(adj + gi + 32, 0, 0);
      float av = adj[gi];
      float e = (av > 0.f) ? lrelu(sF1[r] + f2[j + jj]) : NEG_INF;
      float attv = __expf(e - sM[r]) * sIS[r];
      attbf[idx] = (bf16)attv;
      if (attOut) attOut[gi] = attv;
    }
    __syncthreads();
    v16bf a = cat8(*(const v8bf*)(attbf + l16 * 32 + half * 8),
                   *(const v8bf*)(attbf + l16 * 32 + 16 + half * 8));
    v16bf b = cat8(*(const v8bf*)(brow + j + half * 8),
                   *(const v8bf*)(brow + j + 16 + half * 8));
    c = __builtin_amdgcn_wmma_f32_16x16x32_bf16(false, a, false, b, (short)0, c, false, false);
    __syncthreads();
  }

  int n = col0 + l16;
#pragma unroll
  for (int r = 0; r < 8; ++r) {
    int m = i0 + r + half * 8;
    float v = c[r];
    if (actElu) v = elu1(v);
    if (Hout) Hout[(size_t)m * dout + n] = v;
    Hbf[(size_t)m * dout + n] = (bf16)v;
  }
}

// ---------------------------------------------------------------------------
// genenet = sigmoid(h2 @ h2^T) * rowmask.  One wave per 16x16 tile; K=dh (64).
// h2bf row-major serves as A AND as B (since B = h2^T, B(k,n)=h2[n][k] -> each
// lane's K-gather is contiguous). Tile staged in LDS for coalesced stores.
// ---------------------------------------------------------------------------
__global__ void k_decoder_wmma(const bf16* __restrict__ h2bf, const float* __restrict__ rowmask,
                               float* __restrict__ out, int N, int dh) {
  int wave = blockIdx.x * (blockDim.x >> 5) + (threadIdx.x >> 5);
  int lane = threadIdx.x & 31;
  int tiles = N >> 4;
  int ti = wave / tiles, tj = wave % tiles;
  int half = lane >> 4, l16 = lane & 15;
  const bf16* arow = h2bf + (size_t)(ti * 16 + l16) * dh;
  const bf16* brow = h2bf + (size_t)(tj * 16 + l16) * dh;
  v8f c = {};
  for (int k0 = 0; k0 < dh; k0 += 32) {
    v16bf a = cat8(*(const v8bf*)(arow + k0 + half * 8),
                   *(const v8bf*)(arow + k0 + 16 + half * 8));
    v16bf b = cat8(*(const v8bf*)(brow + k0 + half * 8),
                   *(const v8bf*)(brow + k0 + 16 + half * 8));
    c = __builtin_amdgcn_wmma_f32_16x16x32_bf16(false, a, false, b, (short)0, c, false, false);
  }
  __shared__ float tile[8][16][16];
  int w = threadIdx.x >> 5;
#pragma unroll
  for (int r = 0; r < 8; ++r) tile[w][r + half * 8][l16] = c[r];
  __syncthreads();
  float mask = rowmask[ti * 16 + l16];
  const float* src = &tile[w][l16][half * 8];
  float4 o0, o1;
  o0.x = sigmoidf(src[0]) * mask; o0.y = sigmoidf(src[1]) * mask;
  o0.z = sigmoidf(src[2]) * mask; o0.w = sigmoidf(src[3]) * mask;
  o1.x = sigmoidf(src[4]) * mask; o1.y = sigmoidf(src[5]) * mask;
  o1.z = sigmoidf(src[6]) * mask; o1.w = sigmoidf(src[7]) * mask;
  size_t base = (size_t)(ti * 16 + l16) * N + tj * 16 + half * 8;
  *(float4*)(out + base)     = o0;
  *(float4*)(out + base + 4) = o1;
}

// ---------------------------------------------------------------------------
extern "C" void kernel_launch(void* const* d_in, const int* in_sizes, int n_in,
                              void* d_out, int out_size, void* d_ws, size_t ws_size,
                              hipStream_t stream) {
  const float* features = (const float*)d_in[0];
  const float* adj      = (const float*)d_in[1];
  const float* W1       = (const float*)d_in[2];
  const float* a1       = (const float*)d_in[3];
  const float* W2       = (const float*)d_in[4];
  const float* a2       = (const float*)d_in[5];

  const int dhid  = in_sizes[3] / 2;          // 128
  const int din   = in_sizes[2] / dhid;       // 256
  const int dout2 = in_sizes[5] / 2;          // 64
  const int N     = in_sizes[0] / din;        // 8192

  float* out      = (float*)d_out;
  float* h2_out   = out;                                  // [N][dout2]
  float* gene_out = out + (size_t)N * dout2;              // [N][N]
  float* att_out  = gene_out + (size_t)N * N;             // [N][N]

  // bump allocator on d_ws (all chunks 256B-aligned -> vector loads aligned)
  char* wsb = (char*)d_ws;
  size_t off = 0;
  auto alloc = [&](size_t bytes) -> void* {
    void* p = wsb + off;
    off = (off + bytes + 255) & ~(size_t)255;
    return p;
  };
  bf16*  featbf = (bf16*) alloc((size_t)N * din  * sizeof(bf16));
  bf16*  W1T    = (bf16*) alloc((size_t)din  * dhid  * sizeof(bf16));
  bf16*  W2T    = (bf16*) alloc((size_t)dhid * dout2 * sizeof(bf16));
  float* Wh1    = (float*)alloc((size_t)N * dhid  * sizeof(float));
  bf16*  Wh1T   = (bf16*) alloc((size_t)N * dhid  * sizeof(bf16));
  bf16*  h1bf   = (bf16*) alloc((size_t)N * dhid  * sizeof(bf16));
  float* Wh2    = (float*)alloc((size_t)N * dout2 * sizeof(float));
  bf16*  Wh2T   = (bf16*) alloc((size_t)N * dout2 * sizeof(bf16));
  bf16*  h2bf   = (bf16*) alloc((size_t)N * dout2 * sizeof(bf16));
  float* f1_1   = (float*)alloc((size_t)N * sizeof(float));
  float* f2_1   = (float*)alloc((size_t)N * sizeof(float));
  float* m1     = (float*)alloc((size_t)N * sizeof(float));
  float* s1     = (float*)alloc((size_t)N * sizeof(float));
  float* f1_2   = (float*)alloc((size_t)N * sizeof(float));
  float* f2_2   = (float*)alloc((size_t)N * sizeof(float));
  float* m2     = (float*)alloc((size_t)N * sizeof(float));
  float* s2     = (float*)alloc((size_t)N * sizeof(float));
  float* rmask  = (float*)alloc((size_t)N * sizeof(float));
  (void)ws_size; (void)n_in; (void)out_size;

  // 1) precision staging
  {
    int n = N * din;
    k_f32_to_bf16<<<(n + 255) / 256, 256, 0, stream>>>(features, featbf, n);
    n = din * dhid;
    k_transpose_to_bf16<<<(n + 255) / 256, 256, 0, stream>>>(W1, W1T, din, dhid);
    n = dhid * dout2;
    k_transpose_to_bf16<<<(n + 255) / 256, 256, 0, stream>>>(W2, W2T, dhid, dout2);
  }

  // 2) Wh1 = features @ W1   (fp32 for f-vectors, bf16-transposed for att GEMM)
  {
    int waves = (N / 16) * (dhid / 16);
    k_gemm_wmma_bf16<<<(waves * 32 + 255) / 256, 256, 0, stream>>>(
        featbf, W1T, Wh1, nullptr, Wh1T, N, din, dhid);
  }
  // 3) f1/f2, row stats (+ rowmask) for layer 1
  k_compute_f<<<(N + 255) / 256, 256, 0, stream>>>(Wh1, a1, f1_1, f2_1, N, dhid);
  k_row_stats<<<N, 256, 0, stream>>>(adj, f1_1, f2_1, m1, s1, rmask, N);

  // 4) layer-1 fused: write attention output + h1 = elu(att @ Wh1) as bf16
  k_gat_att_gemm<<<N / 16, 2 * dhid, 0, stream>>>(
      adj, f1_1, f2_1, m1, s1, Wh1T, att_out, nullptr, h1bf, N, dhid, 1);

  // 5) Wh2 = h1 @ W2
  {
    int waves = (N / 16) * (dout2 / 16);
    k_gemm_wmma_bf16<<<(waves * 32 + 255) / 256, 256, 0, stream>>>(
        h1bf, W2T, Wh2, nullptr, Wh2T, N, dhid, dout2);
  }
  // 6) f1/f2, row stats for layer 2
  k_compute_f<<<(N + 255) / 256, 256, 0, stream>>>(Wh2, a2, f1_2, f2_2, N, dout2);
  k_row_stats<<<N, 256, 0, stream>>>(adj, f1_2, f2_2, m2, s2, nullptr, N);

  // 7) layer-2 fused: h2 = att2 @ Wh2 (no activation) -> d_out + bf16 copy
  k_gat_att_gemm<<<N / 16, 2 * dout2, 0, stream>>>(
      adj, f1_2, f2_2, m2, s2, Wh2T, nullptr, h2_out, h2bf, N, dout2, 0);

  // 8) genenet = sigmoid(h2 @ h2^T) * rowmask
  {
    int waves = (N / 16) * (N / 16);
    k_decoder_wmma<<<waves / 8, 256, 0, stream>>>(h2bf, rmask, gene_out, N, dout2);
  }
}
